// Attention_67284957659466
// MI455X (gfx1250) — compile-verified, matrix operand order checked by reading, not run on hardware
//
#include <hip/hip_runtime.h>
#include <hip/hip_bf16.h>

// ---------------------------------------------------------------------------
// MI455X (gfx1250) attention: all GEMMs via v_wmma_f32_16x16x32_bf16 (wave32).
// Big GEMMs use LDS-staged, double-buffered B tiles filled with CDNA5 async
// global->LDS copies (ASYNCcnt) so WMMA issue overlaps global latency.
// ---------------------------------------------------------------------------

typedef __bf16 bf16_t;
typedef __attribute__((ext_vector_type(16))) __bf16 bf16x16;
typedef __attribute__((ext_vector_type(8)))  __bf16 bf16x8;
typedef __attribute__((ext_vector_type(8)))  float  f32x8;
typedef __attribute__((ext_vector_type(4)))  int    i32x4;

#define B_    16
#define N_    596
#define C_    768
#define H_    12
#define DH_   64
#define NCLS_ 20
#define NP_   608              // N padded to 19*32 for the attn@v K loop
#define BN_   (B_ * N_)        // 9536
#define BH_   (B_ * H_)        // 192
#define C3_   (3 * C_)         // 2304
#define MT_N_ 38               // ceil(596/16)

// d_out: [ out (B,N,C) fp32 | weights (B,H,N,N) fp32 ]
#define OUT_W_OFF ((size_t)BN_ * C_)

// workspace layout (bytes); every region is 16B aligned
#define WS_XB     ((size_t)0)                              // (BN, C)   bf16
#define WS_WQKVT  (WS_XB     + (size_t)BN_ * C_   * 2)     // (3C, C)   bf16 (transposed)
#define WS_WPROJT (WS_WQKVT  + (size_t)C3_ * C_   * 2)     // (C, C)    bf16 (transposed)
#define WS_QB     (WS_WPROJT + (size_t)C_  * C_   * 2)     // (BH,N,DH) bf16
#define WS_KB     (WS_QB     + (size_t)BH_ * N_ * DH_ * 2) // (BH,N,DH) bf16
#define WS_VT     (WS_KB     + (size_t)BH_ * N_ * DH_ * 2) // (BH,DH,NP) bf16 (v transposed)
#define WS_ATTNB  (WS_VT     + (size_t)BH_ * DH_ * NP_ * 2)// (BH,N,NP) bf16 (K-padded attn)
#define WS_AOUT   (WS_ATTNB  + (size_t)BH_ * N_ * NP_ * 2) // (BN, C)   bf16

// ---- CDNA5 async global->LDS copy (ASYNCcnt) with portable fallback --------
#if defined(__has_builtin)
#if __has_builtin(__builtin_amdgcn_global_load_async_to_lds_b128)
#define HAVE_ASYNC_LDS 1
#endif
#endif

typedef __attribute__((address_space(1))) i32x4 gvec_t;
typedef __attribute__((address_space(3))) i32x4 lvec_t;

__device__ __forceinline__ void cp16_g2l(const bf16_t* g, bf16_t* l) {
#ifdef HAVE_ASYNC_LDS
    __builtin_amdgcn_global_load_async_to_lds_b128((gvec_t*)(g), (lvec_t*)(l), 0, 0);
#else
    *(bf16x8*)l = *(const bf16x8*)g;
#endif
}

__device__ __forceinline__ void wait_async_copies() {
#ifdef HAVE_ASYNC_LDS
#if __has_builtin(__builtin_amdgcn_s_wait_asynccnt)
    __builtin_amdgcn_s_wait_asynccnt(0);
#else
    asm volatile("s_wait_asynccnt 0x0" ::: "memory");
#endif
#endif
}

// LDS B-tile: 64 rows x 32 bf16, row stride 40 elems (80B = 20 banks; with
// gcd(20,64)=4 the 16 rows a half-wave touches land on 16 distinct phases).
#define LROW_ 40

// cooperative copy: 128 threads move 64 rows x 64B (tid>>1 = row, tid&1 = half)
__device__ __forceinline__ void copy_btile(bf16_t* lbuf, const bf16_t* base,
                                           size_t strideK, int k0, int tid) {
    const int row  = tid >> 1;
    const int half = tid & 1;
    const bf16_t* g = base + (size_t)row * strideK + k0 + half * 16;
    bf16_t* l = lbuf + row * LROW_ + half * 16;
    cp16_g2l(g, l);
    cp16_g2l(g + 8, l + 8);
}

// A-fragment loader (global, row-major, contiguous K): lane l holds row (l&15);
// element j: k = k0 + ((l>=16)?8:0) + (j<8 ? j : j+8)  => two b128 loads.
__device__ __forceinline__ bf16x16 load_frag(const bf16_t* rowbase, int k0, int lhi) {
    union { bf16x16 f; bf16x8 h[2]; } u;
    const bf16_t* p = rowbase + k0 + lhi;
    u.h[0] = *(const bf16x8*)(p);
    u.h[1] = *(const bf16x8*)(p + 16);
    return u.f;
}

// B-fragment from LDS tile (two ds_load_b128)
__device__ __forceinline__ bf16x16 lds_frag(const bf16_t* buf, int row, int lhi) {
    union { bf16x16 f; bf16x8 h[2]; } u;
    const bf16_t* p = buf + row * LROW_ + lhi;
    u.h[0] = *(const bf16x8*)(p);
    u.h[1] = *(const bf16x8*)(p + 16);
    return u.f;
}

#define WMMA(acc, a, b) \
    acc = __builtin_amdgcn_wmma_f32_16x16x32_bf16(false, (a), false, (b), (short)0, (acc), false, false)

// ---------------------------------------------------------------------------
// Kernel 0: bf16 conversion / weight transposes / vT pad zeroing
// ---------------------------------------------------------------------------
#define PREP_R0 ((size_t)BN_ * C_)
#define PREP_R1 (PREP_R0 + (size_t)C3_ * C_)
#define PREP_R2 (PREP_R1 + (size_t)C_ * C_)
#define PREP_R3 (PREP_R2 + (size_t)BH_ * DH_ * (NP_ - N_))

__global__ void __launch_bounds__(256) k_prep(const float* __restrict__ x,
                                              const float* __restrict__ wqkv,
                                              const float* __restrict__ wproj,
                                              bf16_t* __restrict__ xb,
                                              bf16_t* __restrict__ wqkvT,
                                              bf16_t* __restrict__ wprojT,
                                              bf16_t* __restrict__ vT) {
    size_t i = (size_t)blockIdx.x * 256 + threadIdx.x;
    if (i < PREP_R0) {
        xb[i] = (bf16_t)x[i];
    } else if (i < PREP_R1) {
        size_t t = i - PREP_R0;
        size_t n = t / C_, k = t - n * C_;
        wqkvT[n * C_ + k] = (bf16_t)wqkv[k * C3_ + n];
    } else if (i < PREP_R2) {
        size_t t = i - PREP_R1;
        size_t n = t / C_, k = t - n * C_;
        wprojT[n * C_ + k] = (bf16_t)wproj[k * C_ + n];
    } else if (i < PREP_R3) {
        size_t t = i - PREP_R2;
        size_t bhd = t / (NP_ - N_);
        size_t col = N_ + (t - bhd * (NP_ - N_));
        vT[bhd * NP_ + col] = (bf16_t)0.0f;
    }
}

// ---------------------------------------------------------------------------
// Kernel 1: qkv GEMM (9536 x 2304 x 768) + bias -> scatter q / k / vT (bf16)
// block 128 = 4 waves; block tile 128 rows x 64 cols; wave tile 32x64.
// B tile (64x32) LDS double-buffered via async copies. grid (36, 75)
// ---------------------------------------------------------------------------
__global__ void __launch_bounds__(128) k_qkv(const bf16_t* __restrict__ xb,
                                             const bf16_t* __restrict__ wqkvT,
                                             const float* __restrict__ bqkv,
                                             bf16_t* __restrict__ qb,
                                             bf16_t* __restrict__ kb,
                                             bf16_t* __restrict__ vT) {
    __shared__ __align__(16) bf16_t lbuf[2][64 * LROW_];

    const int tid  = threadIdx.x;
    const int lane = tid & 31;
    const int wave = tid >> 5;
    const int ln   = lane & 15;
    const int lhi  = (lane & 16) ? 8 : 0;
    const int n0   = blockIdx.x * 64;
    const int wrow = blockIdx.y * 128 + wave * 32;

    int ar0 = wrow + ln;      if (ar0 > BN_ - 1) ar0 = BN_ - 1;
    int ar1 = wrow + 16 + ln; if (ar1 > BN_ - 1) ar1 = BN_ - 1;
    const bf16_t* arow0 = xb + (size_t)ar0 * C_;
    const bf16_t* arow1 = xb + (size_t)ar1 * C_;
    const bf16_t* bbase = wqkvT + (size_t)n0 * C_;

    const int NC = C_ / 32;                      // 24
    copy_btile(lbuf[0], bbase, C_, 0, tid);
    wait_async_copies();
    __syncthreads();

    f32x8 acc[2][4] = {};
    for (int kc = 0; kc < NC; ++kc) {
        const int cur = kc & 1;
        if (kc + 1 < NC) copy_btile(lbuf[1 - cur], bbase, C_, (kc + 1) * 32, tid);
        bf16x16 a0 = load_frag(arow0, kc * 32, lhi);
        bf16x16 a1 = load_frag(arow1, kc * 32, lhi);
#pragma unroll
        for (int t = 0; t < 4; ++t) {
            bf16x16 b = lds_frag(lbuf[cur], t * 16 + ln, lhi);
            WMMA(acc[0][t], a0, b);
            WMMA(acc[1][t], a1, b);
        }
        wait_async_copies();
        __syncthreads();
    }

#pragma unroll
    for (int u = 0; u < 2; ++u) {
        const int base_m = wrow + u * 16 + ((lane & 16) ? 8 : 0);
#pragma unroll
        for (int r = 0; r < 8; ++r) {
            int m = base_m + r;
            if (m >= BN_) continue;
            int b  = m / N_;
            int nn = m - b * N_;
#pragma unroll
            for (int t = 0; t < 4; ++t) {
                int j   = n0 + t * 16 + ln;
                int s   = j / C_;
                int rem = j - s * C_;
                int h   = rem >> 6, d = rem & 63;
                int bh  = b * H_ + h;
                bf16_t v = (bf16_t)(acc[u][t][r] + bqkv[j]);
                if (s == 0)      qb[((size_t)bh * N_ + nn) * DH_ + d] = v;
                else if (s == 1) kb[((size_t)bh * N_ + nn) * DH_ + d] = v;
                else             vT[((size_t)bh * DH_ + d) * NP_ + nn] = v;
            }
        }
    }
}

// ---------------------------------------------------------------------------
// Kernel 2: scores S = q @ k^T * scale -> fp32 weights region of d_out
// K=64 (2 chunks) -> direct global, wave tile 16x64. grid (10, 10, 192)
// ---------------------------------------------------------------------------
__global__ void __launch_bounds__(128) k_scores(const bf16_t* __restrict__ qb,
                                                const bf16_t* __restrict__ kb,
                                                float* __restrict__ W) {
    const int bh = blockIdx.z;
    const int mt = blockIdx.y * 4 + (threadIdx.x >> 5);
    if (mt >= MT_N_) return;
    const int lane = threadIdx.x & 31;
    const int m0 = mt * 16, n0 = blockIdx.x * 64;
    const int lhi = (lane & 16) ? 8 : 0;
    const int ln  = lane & 15;

    int qr = m0 + ln; if (qr > N_ - 1) qr = N_ - 1;
    const bf16_t* arow = qb + ((size_t)bh * N_ + qr) * DH_;
    const bf16_t* brow[4];
#pragma unroll
    for (int t = 0; t < 4; ++t) {
        int kc = n0 + t * 16 + ln; if (kc > N_ - 1) kc = N_ - 1;
        brow[t] = kb + ((size_t)bh * N_ + kc) * DH_;
    }

    f32x8 acc[4] = {};
#pragma unroll
    for (int k0 = 0; k0 < DH_; k0 += 32) {
        bf16x16 a = load_frag(arow, k0, lhi);
#pragma unroll
        for (int t = 0; t < 4; ++t) {
            bf16x16 b = load_frag(brow[t], k0, lhi);
            WMMA(acc[t], a, b);
        }
    }

    const float scale = 0.125f;
    float* Wbh = W + (size_t)bh * N_ * N_;
    const int base_m = m0 + ((lane & 16) ? 8 : 0);
#pragma unroll
    for (int r = 0; r < 8; ++r) {
        int m = base_m + r;
        if (m >= N_) continue;
#pragma unroll
        for (int t = 0; t < 4; ++t) {
            int n = n0 + t * 16 + ln;
            if (n < N_) Wbh[(size_t)m * N_ + n] = acc[t][r] * scale;
        }
    }
}

// ---------------------------------------------------------------------------
// Kernel 3: dual-group softmax (cols [0,20) and [20,596) separately).
// fp32 back to d_out + bf16 K-padded row into attnb. One block per row.
// ---------------------------------------------------------------------------
__global__ void __launch_bounds__(256) k_softmax(float* __restrict__ W,
                                                 bf16_t* __restrict__ attnb) {
    const int row = blockIdx.x;                    // bh*N + q
    float*  S = W + (size_t)row * N_;
    bf16_t* A = attnb + (size_t)row * NP_;
    const int tid = threadIdx.x;

    float v[3];
#pragma unroll
    for (int i = 0; i < 3; ++i) {
        int c = tid + i * 256;
        v[i] = (c < N_) ? S[c] : -3.0e38f;
    }
    float m0 = -3.0e38f, m1 = -3.0e38f;
#pragma unroll
    for (int i = 0; i < 3; ++i) {
        int c = tid + i * 256;
        if (c < N_) { if (c < NCLS_) m0 = fmaxf(m0, v[i]); else m1 = fmaxf(m1, v[i]); }
    }
    __shared__ float r0[256], r1[256];
    r0[tid] = m0; r1[tid] = m1; __syncthreads();
    for (int s = 128; s > 0; s >>= 1) {
        if (tid < s) { r0[tid] = fmaxf(r0[tid], r0[tid + s]); r1[tid] = fmaxf(r1[tid], r1[tid + s]); }
        __syncthreads();
    }
    m0 = r0[0]; m1 = r1[0]; __syncthreads();

    float e[3], s0 = 0.f, s1 = 0.f;
#pragma unroll
    for (int i = 0; i < 3; ++i) {
        int c = tid + i * 256;
        e[i] = 0.f;
        if (c < N_) {
            bool g = (c < NCLS_);
            e[i] = __expf(v[i] - (g ? m0 : m1));
            if (g) s0 += e[i]; else s1 += e[i];
        }
    }
    r0[tid] = s0; r1[tid] = s1; __syncthreads();
    for (int s = 128; s > 0; s >>= 1) {
        if (tid < s) { r0[tid] += r0[tid + s]; r1[tid] += r1[tid + s]; }
        __syncthreads();
    }
    const float i0 = 1.0f / r0[0], i1 = 1.0f / r1[0];
#pragma unroll
    for (int i = 0; i < 3; ++i) {
        int c = tid + i * 256;
        if (c < N_) {
            float w = e[i] * ((c < NCLS_) ? i0 : i1);
            S[c] = w;
            A[c] = (bf16_t)w;
        }
    }
    if (tid < NP_ - N_) A[N_ + tid] = (bf16_t)0.0f;   // exact zero K-padding
}

// ---------------------------------------------------------------------------
// Kernel 4: out = attn @ v  (M=596 pad 640, N=64, K=608) -> (B,N,C) bf16
// block tile 128 rows x 64 cols, wave tile 32x64, B=vT LDS double-buffered.
// grid (1, 5, 192)
// ---------------------------------------------------------------------------
__global__ void __launch_bounds__(128) k_av(const bf16_t* __restrict__ attnb,
                                            const bf16_t* __restrict__ vT,
                                            bf16_t* __restrict__ aout) {
    __shared__ __align__(16) bf16_t lbuf[2][64 * LROW_];

    const int tid  = threadIdx.x;
    const int lane = tid & 31;
    const int wave = tid >> 5;
    const int ln   = lane & 15;
    const int lhi  = (lane & 16) ? 8 : 0;
    const int bh   = blockIdx.z;
    const int b    = bh / H_, h = bh - b * H_;
    const int wrow = blockIdx.y * 128 + wave * 32;

    int ar0 = wrow + ln;      if (ar0 > N_ - 1) ar0 = N_ - 1;
    int ar1 = wrow + 16 + ln; if (ar1 > N_ - 1) ar1 = N_ - 1;
    const bf16_t* arow0 = attnb + ((size_t)bh * N_ + ar0) * NP_;
    const bf16_t* arow1 = attnb + ((size_t)bh * N_ + ar1) * NP_;
    const bf16_t* bbase = vT + (size_t)bh * DH_ * NP_;   // 64 rows (d), stride NP_

    const int NC = NP_ / 32;                     // 19
    copy_btile(lbuf[0], bbase, NP_, 0, tid);
    wait_async_copies();
    __syncthreads();

    f32x8 acc[2][4] = {};
    for (int kc = 0; kc < NC; ++kc) {
        const int cur = kc & 1;
        if (kc + 1 < NC) copy_btile(lbuf[1 - cur], bbase, NP_, (kc + 1) * 32, tid);
        bf16x16 a0 = load_frag(arow0, kc * 32, lhi);
        bf16x16 a1 = load_frag(arow1, kc * 32, lhi);
#pragma unroll
        for (int t = 0; t < 4; ++t) {
            bf16x16 bf = lds_frag(lbuf[cur], t * 16 + ln, lhi);
            WMMA(acc[0][t], a0, bf);
            WMMA(acc[1][t], a1, bf);
        }
        wait_async_copies();
        __syncthreads();
    }

#pragma unroll
    for (int u = 0; u < 2; ++u) {
        const int base_m = wrow + u * 16 + ((lane & 16) ? 8 : 0);
#pragma unroll
        for (int r = 0; r < 8; ++r) {
            int m = base_m + r;
            if (m >= N_) continue;
#pragma unroll
            for (int t = 0; t < 4; ++t) {
                int d = t * 16 + ln;
                aout[((size_t)b * N_ + m) * C_ + h * DH_ + d] = (bf16_t)acc[u][t][r];
            }
        }
    }
}

// ---------------------------------------------------------------------------
// Kernel 5: proj GEMM (9536 x 768 x 768) + bias -> fp32 out region of d_out
// same LDS/async structure as k_qkv. grid (12, 75)
// ---------------------------------------------------------------------------
__global__ void __launch_bounds__(128) k_proj(const bf16_t* __restrict__ aout,
                                              const bf16_t* __restrict__ wprojT,
                                              const float* __restrict__ bproj,
                                              float* __restrict__ out) {
    __shared__ __align__(16) bf16_t lbuf[2][64 * LROW_];

    const int tid  = threadIdx.x;
    const int lane = tid & 31;
    const int wave = tid >> 5;
    const int ln   = lane & 15;
    const int lhi  = (lane & 16) ? 8 : 0;
    const int n0   = blockIdx.x * 64;
    const int wrow = blockIdx.y * 128 + wave * 32;

    int ar0 = wrow + ln;      if (ar0 > BN_ - 1) ar0 = BN_ - 1;
    int ar1 = wrow + 16 + ln; if (ar1 > BN_ - 1) ar1 = BN_ - 1;
    const bf16_t* arow0 = aout + (size_t)ar0 * C_;
    const bf16_t* arow1 = aout + (size_t)ar1 * C_;
    const bf16_t* bbase = wprojT + (size_t)n0 * C_;

    const int NC = C_ / 32;                      // 24
    copy_btile(lbuf[0], bbase, C_, 0, tid);
    wait_async_copies();
    __syncthreads();

    f32x8 acc[2][4] = {};
    for (int kc = 0; kc < NC; ++kc) {
        const int cur = kc & 1;
        if (kc + 1 < NC) copy_btile(lbuf[1 - cur], bbase, C_, (kc + 1) * 32, tid);
        bf16x16 a0 = load_frag(arow0, kc * 32, lhi);
        bf16x16 a1 = load_frag(arow1, kc * 32, lhi);
#pragma unroll
        for (int t = 0; t < 4; ++t) {
            bf16x16 b = lds_frag(lbuf[cur], t * 16 + ln, lhi);
            WMMA(acc[0][t], a0, b);
            WMMA(acc[1][t], a1, b);
        }
        wait_async_copies();
        __syncthreads();
    }

#pragma unroll
    for (int u = 0; u < 2; ++u) {
        const int base_m = wrow + u * 16 + ((lane & 16) ? 8 : 0);
#pragma unroll
        for (int r = 0; r < 8; ++r) {
            int m = base_m + r;
            if (m >= BN_) continue;
#pragma unroll
            for (int t = 0; t < 4; ++t) {
                int n = n0 + t * 16 + ln;
                out[(size_t)m * C_ + n] = acc[u][t][r] + bproj[n];
            }
        }
    }
}

// ---------------------------------------------------------------------------
extern "C" void kernel_launch(void* const* d_in, const int* in_sizes, int n_in,
                              void* d_out, int out_size, void* d_ws, size_t ws_size,
                              hipStream_t stream) {
    const float* x     = (const float*)d_in[0];
    const float* wqkv  = (const float*)d_in[1];
    const float* bqkv  = (const float*)d_in[2];
    const float* wproj = (const float*)d_in[3];
    const float* bproj = (const float*)d_in[4];

    float* out = (float*)d_out;                 // (B,N,C)
    float* W   = out + OUT_W_OFF;               // (B,H,N,N) weights

    char* ws = (char*)d_ws;
    bf16_t* xb     = (bf16_t*)(ws + WS_XB);
    bf16_t* wqkvT  = (bf16_t*)(ws + WS_WQKVT);
    bf16_t* wprojT = (bf16_t*)(ws + WS_WPROJT);
    bf16_t* qb     = (bf16_t*)(ws + WS_QB);
    bf16_t* kb     = (bf16_t*)(ws + WS_KB);
    bf16_t* vT     = (bf16_t*)(ws + WS_VT);
    bf16_t* attnb  = (bf16_t*)(ws + WS_ATTNB);
    bf16_t* aout   = (bf16_t*)(ws + WS_AOUT);

    k_prep   <<<dim3((unsigned)((PREP_R3 + 255) / 256)), 256, 0, stream>>>(x, wqkv, wproj, xb, wqkvT, wprojT, vT);
    k_qkv    <<<dim3(C3_ / 64, (BN_ + 127) / 128, 1), 128, 0, stream>>>(xb, wqkvT, bqkv, qb, kb, vT);
    k_scores <<<dim3(10, 10, BH_), 128, 0, stream>>>(qb, kb, W);
    k_softmax<<<dim3(BH_ * N_), 256, 0, stream>>>(W, attnb);
    k_av     <<<dim3(1, 5, BH_), 128, 0, stream>>>(attnb, vT, aout);
    k_proj   <<<dim3(C_ / 64, (BN_ + 127) / 128, 1), 128, 0, stream>>>(aout, wprojT, bproj, out);
}
